// CrossModalFusion_54150947668621
// MI455X (gfx1250) — compile-verified, matrix operand order checked by reading, not run on hardware
//
#include <hip/hip_runtime.h>
#include <hip/hip_bf16.h>

typedef __attribute__((ext_vector_type(8)))  float  v8f;
typedef __attribute__((ext_vector_type(8)))  __bf16 v8bf;
typedef __attribute__((ext_vector_type(16))) __bf16 v16bf;

#define H_DIM  1024
#define B_ROWS 16384
#define TILE_M 128
#define TILE_K 32
#define LDSP   40   // padded LDS row stride in halves (80B -> spreads banks)

__device__ __forceinline__ unsigned short f2bf(float f) {
  unsigned int u = __float_as_uint(f);
  u += 0x7fffu + ((u >> 16) & 1u);   // round-to-nearest-even
  return (unsigned short)(u >> 16);
}

__device__ __forceinline__ unsigned lds_off32(const void* p) {
  // generic LDS pointer: low 32 bits hold the LDS byte offset
  return (unsigned)(unsigned long long)p;
}

// async copy 16B: global -> LDS (per lane), tracked by ASYNCcnt
#define ASYNC_LD16(ldsoff, gptr)                                             \
  asm volatile("global_load_async_to_lds_b128 %0, %1, off"                   \
               :: "v"(ldsoff), "v"((unsigned long long)(const void*)(gptr))  \
               : "memory")

__global__ void cvt_bf16x4_kernel(const float4* __restrict__ src,
                                  ushort4* __restrict__ dst, long long n4) {
  long long i = (long long)blockIdx.x * blockDim.x + threadIdx.x;
  const long long stride = (long long)gridDim.x * blockDim.x;
  for (; i < n4; i += stride) {
    const float4 f = src[i];
    ushort4 o;
    o.x = f2bf(f.x); o.y = f2bf(f.y); o.z = f2bf(f.z); o.w = f2bf(f.w);
    dst[i] = o;
  }
}

// C[M,N] = act(A[M,K](bf16) * W[N,K]^T(bf16) [+bias] [+resid]); f32 or bf16 out.
// Block tile: 128 x BN, 8 waves; wave tile 32 x (BN/2); K stepped by 32 through
// double-buffered LDS filled with global_load_async_to_lds_b128.
template<int KK, int NN, int BN, bool HAS_BIAS, bool HAS_RESID, bool RELU, bool OUT_BF16>
__global__ __launch_bounds__(256) void gemm_bf16_kernel(
    const unsigned short* __restrict__ A,
    const unsigned short* __restrict__ W,
    const float* __restrict__ bias,
    const float* __restrict__ resid,
    float* __restrict__ Cf,
    unsigned short* __restrict__ Cbf)
{
  constexpr int NJ  = BN / 32;          // 16-col WMMA tiles per wave (N dir)
  constexpr int BCH = (BN * 4) / 256;   // B staging chunks per thread (1 or 2)
  constexpr int KT  = KK / TILE_K;
  __shared__ unsigned short As[2][TILE_M * LDSP];
  __shared__ unsigned short Bs[2][BN * LDSP];

  const int tid  = threadIdx.x;
  const int lane = tid & 31;
  const int wave = tid >> 5;            // 0..7
  const int wm   = wave >> 1;           // 0..3 (M)
  const int wn   = wave & 1;            // 0..1 (N)
  const int tileM = blockIdx.y * TILE_M;
  const int tileN = blockIdx.x * BN;
  const int l  = lane & 15;
  const int hi = lane >> 4;
  const int c0 = hi * 8;                // halves: first contiguous K chunk
  const int c1 = 16 + hi * 8;           // halves: second contiguous K chunk

  v8f acc[2][NJ] = {};

  // ---- staging assignment: 16B chunks, 4 per row of 32 halves
  const int r0 = tid >> 2;              // 0..63
  const int r1 = r0 + 64;               // 64..127
  const int s0 = (tid & 3) * 8;         // halves offset within k-tile row

  const unsigned short* aP0 = A + (size_t)(tileM + r0) * KK + s0;
  const unsigned short* aP1 = A + (size_t)(tileM + r1) * KK + s0;
  const unsigned short* bP0 = W + (size_t)(tileN + r0) * KK + s0;
  const unsigned short* bP1 = (BCH == 2) ? (W + (size_t)(tileN + r1) * KK + s0) : nullptr;

  unsigned ldsA0[2], ldsA1[2], ldsB0[2], ldsB1[2];
#pragma unroll
  for (int b = 0; b < 2; ++b) {
    ldsA0[b] = lds_off32(&As[b][r0 * LDSP + s0]);
    ldsA1[b] = lds_off32(&As[b][r1 * LDSP + s0]);
    ldsB0[b] = lds_off32(&Bs[b][r0 * LDSP + s0]);
    if (BCH == 2) ldsB1[b] = lds_off32(&Bs[b][r1 * LDSP + s0]);
  }

  // prologue: fill buffer 0
  ASYNC_LD16(ldsA0[0], aP0);
  ASYNC_LD16(ldsA1[0], aP1);
  ASYNC_LD16(ldsB0[0], bP0);
  if (BCH == 2) ASYNC_LD16(ldsB1[0], bP1);
  asm volatile("s_wait_asynccnt 0x0" ::: "memory");
  __syncthreads();

#pragma unroll 2
  for (int kt = 0; kt < KT; ++kt) {
    const int cur = kt & 1;
    const int nxt = cur ^ 1;
    // unconditional: one-past-end tile reads stay inside the workspace and
    // are never consumed.
    {
      const int ko = (kt + 1) * TILE_K;
      ASYNC_LD16(ldsA0[nxt], aP0 + ko);
      ASYNC_LD16(ldsA1[nxt], aP1 + ko);
      ASYNC_LD16(ldsB0[nxt], bP0 + ko);
      if (BCH == 2) ASYNC_LD16(ldsB1[nxt], bP1 + ko);
    }

    v16bf afrag[2], bfrag[NJ];
#pragma unroll
    for (int i = 0; i < 2; ++i) {
      const int row = wm * 32 + i * 16 + l;
      v8bf lo = *(const v8bf*)(const void*)(&As[cur][row * LDSP + c0]);
      v8bf hh = *(const v8bf*)(const void*)(&As[cur][row * LDSP + c1]);
      afrag[i] = __builtin_shufflevector(lo, hh, 0,1,2,3,4,5,6,7,8,9,10,11,12,13,14,15);
    }
#pragma unroll
    for (int j = 0; j < NJ; ++j) {
      const int row = wn * (BN / 2) + j * 16 + l;
      v8bf lo = *(const v8bf*)(const void*)(&Bs[cur][row * LDSP + c0]);
      v8bf hh = *(const v8bf*)(const void*)(&Bs[cur][row * LDSP + c1]);
      bfrag[j] = __builtin_shufflevector(lo, hh, 0,1,2,3,4,5,6,7,8,9,10,11,12,13,14,15);
    }
#pragma unroll
    for (int i = 0; i < 2; ++i)
#pragma unroll
      for (int j = 0; j < NJ; ++j)
        acc[i][j] = __builtin_amdgcn_wmma_f32_16x16x32_bf16(
            false, afrag[i], false, bfrag[j], (short)0, acc[i][j], false, false);

    asm volatile("s_wait_asynccnt 0x0" ::: "memory");
    __syncthreads();
  }

  // ---- epilogue: C/D layout -> lanes 0-15: M=r, lanes 16-31: M=8+r, N=l.
  // All offsets except `base`/`colBase` are compile-time constants.
  const unsigned colBase = tileN + wn * (BN / 2) + l;
  const unsigned base    = (unsigned)(tileM + wm * 32 + hi * 8) * NN + colBase;
#pragma unroll
  for (int i = 0; i < 2; ++i) {
#pragma unroll
    for (int j = 0; j < NJ; ++j) {
      float cb = 0.0f;
      if constexpr (HAS_BIAS) cb = bias[colBase + j * 16];
#pragma unroll
      for (int r = 0; r < 8; ++r) {
        float v = acc[i][j][r] + cb;
        const unsigned off = base + (unsigned)(i * 16 * NN + r * NN + j * 16);
        if constexpr (HAS_RESID) v += resid[off];
        if constexpr (RELU) v = fmaxf(v, 0.0f);
        if constexpr (OUT_BF16) Cbf[off] = f2bf(v);
        else                    Cf[off]  = v;
      }
    }
  }
  (void)bias; (void)resid; (void)Cf; (void)Cbf;
}

// In-place LayerNorm over H=1024 per row; also emits a bf16 copy.
__global__ __launch_bounds__(256) void ln_kernel(float* __restrict__ Z,
    const float* __restrict__ g, const float* __restrict__ b,
    unsigned short* __restrict__ Ybf)
{
  __shared__ float red[256];
  const int row = blockIdx.x;
  const int tid = threadIdx.x;
  const size_t base = (size_t)row * H_DIM;
  float x[4];
  float s = 0.f;
#pragma unroll
  for (int i = 0; i < 4; ++i) { x[i] = Z[base + tid + i * 256]; s += x[i]; }
  red[tid] = s; __syncthreads();
  for (int o = 128; o > 0; o >>= 1) { if (tid < o) red[tid] += red[tid + o]; __syncthreads(); }
  const float mean = red[0] * (1.0f / H_DIM);
  __syncthreads();
  float s2 = 0.f;
#pragma unroll
  for (int i = 0; i < 4; ++i) { const float d = x[i] - mean; s2 += d * d; }
  red[tid] = s2; __syncthreads();
  for (int o = 128; o > 0; o >>= 1) { if (tid < o) red[tid] += red[tid + o]; __syncthreads(); }
  const float rstd = rsqrtf(red[0] * (1.0f / H_DIM) + 1e-5f);
#pragma unroll
  for (int i = 0; i < 4; ++i) {
    const int c = tid + i * 256;
    const float y = (x[i] - mean) * rstd * g[c] + b[c];
    Z[base + c]   = y;
    Ybf[base + c] = f2bf(y);
  }
}

// x2 = y * 2*sigmoid(S2); sa = conv7(x2) with combined 2-ch weights; out = x2*sigmoid(sa)
__global__ __launch_bounds__(256) void cbam_kernel(const float* __restrict__ Y,
    const float* __restrict__ S2, const float* __restrict__ conv,
    float* __restrict__ Out)
{
  __shared__ float x2[H_DIM];
  __shared__ float wk[7];
  const int row = blockIdx.x, tid = threadIdx.x;
  if (tid < 7) wk[tid] = conv[tid] + conv[7 + tid];  // both input channels equal
  const size_t base = (size_t)row * H_DIM;
#pragma unroll
  for (int i = 0; i < 4; ++i) {
    const int c = tid + i * 256;
    const float ca = 1.0f / (1.0f + expf(-S2[base + c]));
    x2[c] = Y[base + c] * 2.0f * ca;
  }
  __syncthreads();
#pragma unroll
  for (int i = 0; i < 4; ++i) {
    const int c = tid + i * 256;
    float sa = 0.f;
#pragma unroll
    for (int k = 0; k < 7; ++k) {
      const int p = c + k - 3;
      const float xv = (p >= 0 && p < H_DIM) ? x2[p] : 0.0f;
      sa += wk[k] * xv;
    }
    Out[base + c] = x2[c] * (1.0f / (1.0f + expf(-sa)));
  }
}

__global__ void min4_kernel(float4* __restrict__ out, const float4* __restrict__ oth,
                            long long n4) {
  long long i = (long long)blockIdx.x * blockDim.x + threadIdx.x;
  const long long stride = (long long)gridDim.x * blockDim.x;
  for (; i < n4; i += stride) {
    float4 a = out[i];
    const float4 b = oth[i];
    a.x = fminf(a.x, b.x); a.y = fminf(a.y, b.y);
    a.z = fminf(a.z, b.z); a.w = fminf(a.w, b.w);
    out[i] = a;
  }
}

extern "C" void kernel_launch(void* const* d_in, const int* in_sizes, int n_in,
                              void* d_out, int out_size, void* d_ws, size_t ws_size,
                              hipStream_t stream)
{
  (void)in_sizes; (void)n_in; (void)out_size; (void)ws_size;
  const size_t BH = (size_t)B_ROWS * H_DIM;
  const size_t HH = (size_t)H_DIM * H_DIM;
  const size_t HR = (size_t)H_DIM * 64;

  // ---- workspace carve (all chunk sizes multiples of 8 ushorts -> 16B aligned)
  unsigned short* p = (unsigned short*)d_ws;
  unsigned short* bf_text  = p; p += BH;
  unsigned short* bf_audio = p; p += BH;
  unsigned short* bf_t1    = p; p += BH;
  unsigned short* bf_t2    = p; p += BH;
  unsigned short* s1_bf    = p; p += (size_t)B_ROWS * 64;
  unsigned short* wbf[2][5];
  for (int br = 0; br < 2; ++br) {
    wbf[br][0] = p; p += HH;  // Wv
    wbf[br][1] = p; p += HH;  // Wo
    wbf[br][2] = p; p += HH;  // lin_W
    wbf[br][3] = p; p += HR;  // cb_W1 [64,1024]
    wbf[br][4] = p; p += HR;  // cb_W2 [1024,64]
  }
  float* f32P = (float*)p;    // Z -> y (LayerNorm in place)
  float* f32Q = f32P + BH;    // S2
  float* f32R = f32Q + BH;    // branch-a output

  auto cvt = [&](const void* s, unsigned short* d, size_t n) {
    const long long n4 = (long long)(n / 4);
    int blocks = (int)((n4 + 255) / 256); if (blocks > 4096) blocks = 4096;
    cvt_bf16x4_kernel<<<blocks, 256, 0, stream>>>(
        (const float4*)s, (ushort4*)d, n4);
  };

  cvt(d_in[0], bf_text,  BH);
  cvt(d_in[1], bf_audio, BH);
  for (int br = 0; br < 2; ++br) {
    const int ib = 2 + br * 11;
    cvt(d_in[ib + 0], wbf[br][0], HH);   // Wv
    cvt(d_in[ib + 2], wbf[br][1], HH);   // Wo
    cvt(d_in[ib + 4], wbf[br][2], HH);   // lin_W
    cvt(d_in[ib + 8], wbf[br][3], HR);   // cb_W1
    cvt(d_in[ib + 9], wbf[br][4], HR);   // cb_W2
  }

  auto branch = [&](const unsigned short* kv_bf, const float* query, int ib, float* out) {
    const float* bv    = (const float*)d_in[ib + 1];
    const float* bo    = (const float*)d_in[ib + 3];
    const float* lin_b = (const float*)d_in[ib + 5];
    const float* ln_g  = (const float*)d_in[ib + 6];
    const float* ln_b  = (const float*)d_in[ib + 7];
    const float* convw = (const float*)d_in[ib + 10];
    const int br = (ib == 2) ? 0 : 1;
    const dim3 gBig(H_DIM / 128, B_ROWS / TILE_M);   // N=1024 tiles
    const dim3 gRed(64 / 64,     B_ROWS / TILE_M);   // N=64 tiles

    // T1 = kv Wv^T + bv          (bf16 out)
    gemm_bf16_kernel<1024, 1024, 128, true, false, false, true><<<gBig, 256, 0, stream>>>(
        kv_bf, wbf[br][0], bv, nullptr, nullptr, bf_t1);
    // attn = T1 Wo^T + bo        (bf16 out)
    gemm_bf16_kernel<1024, 1024, 128, true, false, false, true><<<gBig, 256, 0, stream>>>(
        bf_t1, wbf[br][1], bo, nullptr, nullptr, bf_t2);
    // Z = attn lin_W^T + lin_b + query   (f32 out)
    gemm_bf16_kernel<1024, 1024, 128, true, true, false, false><<<gBig, 256, 0, stream>>>(
        bf_t2, wbf[br][2], lin_b, query, f32P, nullptr);
    // y = LN(Z) in place; bf16 copy in bf_t1
    ln_kernel<<<B_ROWS, 256, 0, stream>>>(f32P, ln_g, ln_b, bf_t1);
    // S1 = relu(y W1^T)          (bf16 out, N=64)
    gemm_bf16_kernel<1024, 64, 64, false, false, true, true><<<gRed, 256, 0, stream>>>(
        bf_t1, wbf[br][3], nullptr, nullptr, nullptr, s1_bf);
    // S2 = S1 W2^T               (f32 out, K=64)
    gemm_bf16_kernel<64, 1024, 128, false, false, false, false><<<gBig, 256, 0, stream>>>(
        s1_bf, wbf[br][4], nullptr, nullptr, f32Q, nullptr);
    // CBAM scale + spatial conv
    cbam_kernel<<<B_ROWS, 256, 0, stream>>>(f32P, f32Q, convw, out);
  };

  branch(bf_audio, (const float*)d_in[0], 2,  f32R);            // audio_attn (query=text)
  branch(bf_text,  (const float*)d_in[1], 13, (float*)d_out);   // text_attn (query=audio)
  min4_kernel<<<4096, 256, 0, stream>>>((float4*)d_out, (const float4*)f32R,
                                        (long long)(BH / 4));
}
